// VoxelBackBone8x_mod1_87780541596066
// MI455X (gfx1250) — compile-verified, matrix operand order checked by reading, not run on hardware
//
#include <hip/hip_runtime.h>
#include <cstdint>

// ---------------------------------------------------------------------------
// Sparse voxel backbone on MI455X (gfx1250, wave32).
//  - activations: f16 NDHWC (C innermost, padded to mult of 16) -> whole
//    stage-1 feature map (86MB) is L2-resident (192MB L2).
//  - convs: implicit GEMM on v_wmma_f32_16x16x32_f16; one block (1 wave)
//    stages a halo tile to LDS once (async-to-LDS), then computes ALL
//    co-tiles for a 16-position x-tile: B built once per k-step as two
//    ds_load_b128 runs, COT wmmas per k-step.
//  - geometry templated -> all div/mod constant-folded.
//  - BN folded into weights on device; epilogue fuses bias+residual+relu+mask.
// ---------------------------------------------------------------------------

#define XT 16  // output x positions per wave tile

typedef __attribute__((ext_vector_type(16))) _Float16 v16h;
typedef __attribute__((ext_vector_type(8)))  _Float16 v8h;
typedef __attribute__((ext_vector_type(8)))  float    v8f;

#if defined(__has_builtin)
#  if __has_builtin(__builtin_amdgcn_global_load_async_to_lds_b128)
#    define HAVE_ASYNC 1
#  else
#    define HAVE_ASYNC 0
#  endif
#  if __has_builtin(__builtin_amdgcn_s_wait_asynccnt)
#    define HAVE_WAITASYNC 1
#  else
#    define HAVE_WAITASYNC 0
#  endif
#else
#  define HAVE_ASYNC 0
#  define HAVE_WAITASYNC 0
#endif

#if HAVE_ASYNC
// builtin expects pointers to int4 vectors in global (addrspace 1) and
// LDS (addrspace 3) address spaces (per hipcc diagnostic).
typedef int v4i_ __attribute__((vector_size(16)));
typedef __attribute__((address_space(1))) v4i_ gv4i;
typedef __attribute__((address_space(3))) v4i_ lv4i;
#endif

// K index held by (lane, f16-slot j) for a 16-bit 16x32 A fragment
// (CDNA5 ISA 7.12.2): j=0..7 -> K = hi8+j, j=8..15 -> K = 16+hi8+(j-8).
// Used by the weight packer; the conv loads B as two contiguous 8-element runs
// that realize the same mapping.
__device__ __forceinline__ int kmap16(int lane, int j) {
  int hi8 = (lane & 16) ? 8 : 0;
  int v = j >> 1;
  int base = (v < 4) ? (v * 2) : (16 + (v - 4) * 2);
  return base + hi8 + (j & 1);
}

__device__ __forceinline__ v8h zero8() {
  v8h z;
#pragma unroll
  for (int i = 0; i < 8; ++i) z[i] = (_Float16)0.0f;
  return z;
}

// ---------------------------------------------------------------------------
// Scatter voxel features into dense f32 accumulation grid + counts
// ---------------------------------------------------------------------------
__global__ __launch_bounds__(256) void scatter_k(const float* vf, const int* vc,
                                                 float* accF, float* cntF,
                                                 int nv, int H, int W) {
  int i = blockIdx.x * 256 + threadIdx.x;
  if (i >= nv) return;
  int z = vc[i * 4 + 1], y = vc[i * 4 + 2], x = vc[i * 4 + 3];
  size_t v = ((size_t)z * H + y) * W + x;
#pragma unroll
  for (int c = 0; c < 4; ++c) atomicAdd(&accF[v * 4 + c], vf[i * 4 + c]);
  atomicAdd(&cntF[v], 1.0f);
}

// dense f32 (C=4) -> f16 NDHWC C=16 (zero padded) + occupancy mask
__global__ __launch_bounds__(256) void build_input_k(const float* accF, const float* cntF,
                                                     _Float16* in0, _Float16* m1, long tot) {
  long t = (long)blockIdx.x * 256 + threadIdx.x;
  if (t >= tot) return;
  m1[t] = (cntF[t] > 0.0f) ? (_Float16)1.0f : (_Float16)0.0f;
#pragma unroll
  for (int ch = 0; ch < 16; ++ch)
    in0[t * 16 + ch] = (ch < 4) ? (_Float16)accF[t * 4 + ch] : (_Float16)0.0f;
}

// max-pool mask with zero padding -> binary mask at output resolution
__global__ __launch_bounds__(256) void down_mask_k(const _Float16* mi, _Float16* mo,
                                                   int D, int H, int W,
                                                   int OD, int OH, int OW,
                                                   int KD, int KH, int KW,
                                                   int sz, int sy, int sx,
                                                   int pz, int py, int px) {
  long t = (long)blockIdx.x * 256 + threadIdx.x;
  long tot = (long)OD * OH * OW;
  if (t >= tot) return;
  int ox = (int)(t % OW);
  int oy = (int)((t / OW) % OH);
  int oz = (int)(t / ((long)OW * OH));
  float m = 0.0f;
  for (int kz = 0; kz < KD; ++kz)
    for (int ky = 0; ky < KH; ++ky)
      for (int kx = 0; kx < KW; ++kx) {
        int iz = oz * sz - pz + kz;
        int iy = oy * sy - py + ky;
        int ix = ox * sx - px + kx;
        if (iz >= 0 && iz < D && iy >= 0 && iy < H && ix >= 0 && ix < W)
          m = fmaxf(m, (float)mi[((size_t)iz * H + iy) * W + ix]);
      }
  mo[t] = (m > 0.0f) ? (_Float16)1.0f : (_Float16)0.0f;
}

// ---------------------------------------------------------------------------
// Weight packing: DHWIO f32 -> f16 A-fragment tiles in exact WMMA lane layout,
// BN scale folded in; K dim = tap*Ci + c (zero for pad taps / pad channels).
// ---------------------------------------------------------------------------
struct PackP {
  const float* w;
  const float* g;
  const float* vr;
  _Float16* wpk;
  int CiR, Ci, ciShift, Co, KT, Ksteps;
};

__global__ __launch_bounds__(256) void pack_w_k(PackP q, long total) {
  long t = (long)blockIdx.x * 256 + threadIdx.x;
  if (t >= total) return;
  int j = (int)(t & 15);
  int lane = (int)((t >> 4) & 31);
  long tile = t >> 9;
  int ks = (int)(tile % q.Ksteps);
  int cot = (int)(tile / q.Ksteps);
  int co = cot * 16 + (lane & 15);
  int k = ks * 32 + kmap16(lane, j);
  int c = k & (q.Ci - 1);
  int tap = k >> q.ciShift;
  float val = 0.0f;
  if (tap < q.KT && c < q.CiR) {
    float s = q.g[co] * rsqrtf(q.vr[co] + 1e-3f);
    val = q.w[(size_t)tap * q.CiR * q.Co + (size_t)c * q.Co + co] * s;
  }
  q.wpk[t] = (_Float16)val;
}

__global__ __launch_bounds__(64) void pack_bias_k(const float* g, const float* be,
                                                  const float* mn, const float* vr,
                                                  const float* b, float* bias, int Co) {
  int co = blockIdx.x * 64 + threadIdx.x;
  if (co >= Co) return;
  float s = g[co] * rsqrtf(vr[co] + 1e-3f);
  float bb = b ? b[co] : 0.0f;
  bias[co] = bb * s + be[co] - mn[co] * s;
}

// ---------------------------------------------------------------------------
// Implicit-GEMM 3D conv. One wave -> (COT*16 co) x (16 ox) tile at fixed
// (oz,oy). Epilogue: out = relu(acc + bias (+ resid)) * mask, stored f16.
// ---------------------------------------------------------------------------
struct ConvP {
  const _Float16* x;
  const _Float16* wpk;
  const float* bias;
  const _Float16* maskO;
  const _Float16* resid;  // nullptr for non-residual convs
  _Float16* y;
  int D, H, W;
  int OH, OW;
  int sz, sy;             // sx is a template parameter
  int pz, py, px;
};

template <int KD, int KH, int KW, int SX, int CI, int COT>
__global__ __launch_bounds__(32) void conv3d_wmma_k(ConvP p) {
  constexpr int SPAN = (XT - 1) * SX + KW;
  constexpr int KT = KD * KH * KW;
  constexpr int KSTEPS = (KT * CI + 31) / 32;
  constexpr int CO = COT * 16;
  constexpr int CISHIFT = (CI == 16) ? 4 : (CI == 32) ? 5 : 6;
  constexpr int CPV = CI >> 3;  // 16B chunks per voxel
  constexpr int NVOX = KD * KH * SPAN;
  constexpr int NCHUNKS = NVOX * CPV;

  extern __shared__ _Float16 smem[];
  const int lane = threadIdx.x;
  const int xtiles = p.OW >> 4;  // all OW in this net are multiples of 16
  long tid = blockIdx.x;
  const int xt = (int)(tid % xtiles);
  long pt = tid / xtiles;
  const int oy = (int)(pt % p.OH);
  const int oz = (int)(pt / p.OH);

  const int x0 = xt * XT * SX - p.px;
  const int y0 = oy * p.sy - p.py;
  const int z0 = oz * p.sz - p.pz;

  // ---- stage halo tile [KD][KH][SPAN][CI] into LDS (16B chunks) ----
  for (int c = lane; c < NCHUNKS; c += 32) {
    int vox = c / CPV;
    int ce = (c - vox * CPV) << 3;  // f16 element offset within voxel
    int xi = vox % SPAN;
    int t2 = vox / SPAN;
    int ky = t2 % KH;
    int kz = t2 / KH;
    int gx = x0 + xi, gy = y0 + ky, gz = z0 + kz;
    _Float16* dst = smem + (size_t)vox * CI + ce;
    if (gx >= 0 && gx < p.W && gy >= 0 && gy < p.H && gz >= 0 && gz < p.D) {
      const _Float16* src = p.x + (((size_t)gz * p.H + gy) * p.W + gx) * CI + ce;
#if HAVE_ASYNC
      __builtin_amdgcn_global_load_async_to_lds_b128(
          (gv4i*)(const void*)src, (lv4i*)(void*)dst, 0, 0);
#else
      *(uint4*)dst = *(const uint4*)src;
#endif
    } else {
      uint4 z4; z4.x = 0u; z4.y = 0u; z4.z = 0u; z4.w = 0u;
      *(uint4*)dst = z4;
    }
  }
#if HAVE_ASYNC
#  if HAVE_WAITASYNC
  __builtin_amdgcn_s_wait_asynccnt(0);
#  else
  asm volatile("s_wait_asynccnt 0" ::: "memory");
#  endif
#endif
  // single wave per workgroup: DS ops are in-order, no barrier needed

  const int hi = (lane >> 4) & 1;
  const int n = lane & 15;

  v8f acc[COT];
#pragma unroll
  for (int ct = 0; ct < COT; ++ct)
#pragma unroll
    for (int r = 0; r < 8; ++r) acc[ct][r] = p.bias[ct * 16 + hi * 8 + r];

  for (int ks = 0; ks < KSTEPS; ++ks) {
    // ---- B fragment: two contiguous 8-f16 runs (ds_load_b128 each) ----
    // run rr covers slots j = rr*8 .. rr*8+7 <-> K = ks*32 + rr*16 + hi*8 + j'
    v8h run0, run1;
#pragma unroll
    for (int rr = 0; rr < 2; ++rr) {
      int K = ks * 32 + rr * 16 + hi * 8;
      int tap = K >> CISHIFT;
      int cch = K & (CI - 1);
      v8h r8;
      if (tap < KT) {  // only divergent on the zero-padded tail k-step
        int kx = tap % KW;
        int t2 = tap / KW;
        int kyy = t2 % KH;
        int kzz = t2 / KH;
        int vox = (kzz * KH + kyy) * SPAN + n * SX + kx;
        r8 = *(const v8h*)(smem + (size_t)vox * CI + cch);
      } else {
        r8 = zero8();
      }
      if (rr == 0) run0 = r8; else run1 = r8;
    }
    v16h b = __builtin_shufflevector(run0, run1,
                                     0, 1, 2, 3, 4, 5, 6, 7,
                                     8, 9, 10, 11, 12, 13, 14, 15);

    // ---- COT wmmas against the shared B fragment ----
#pragma unroll
    for (int ct = 0; ct < COT; ++ct) {
      const _Float16* aptr = p.wpk + (((size_t)ct * KSTEPS + ks) * 32 + lane) * 16;
      v16h a = *(const v16h*)aptr;
      acc[ct] = __builtin_amdgcn_wmma_f32_16x16x32_f16(
          /*neg_a=*/false, a, /*neg_b=*/false, b,
          /*c_mod=*/(short)0, acc[ct], /*reuse_a=*/false, /*reuse_b=*/false);
    }
    if (ks + 1 < KSTEPS)
      __builtin_prefetch(p.wpk + ((size_t)(ks + 1) * 32 + lane) * 16, 0, 1);
  }

  // ---- epilogue: bias already in acc; add resid, relu, mask, store f16 ----
  const int ox = xt * XT + n;
  size_t ovox = ((size_t)oz * p.OH + oy) * p.OW + ox;
  float m = (float)p.maskO[ovox];
#pragma unroll
  for (int ct = 0; ct < COT; ++ct) {
    size_t obase = ovox * CO + ct * 16 + hi * 8;
#pragma unroll
    for (int r = 0; r < 8; r += 2) {
      float v0 = acc[ct][r], v1 = acc[ct][r + 1];
      if (p.resid) {
        v0 += (float)p.resid[obase + r];
        v1 += (float)p.resid[obase + r + 1];
      }
      v0 = fmaxf(v0, 0.0f) * m;
      v1 = fmaxf(v1, 0.0f) * m;
      union { _Float16 h[2]; unsigned u; } pk;
      pk.h[0] = (_Float16)v0;
      pk.h[1] = (_Float16)v1;
      *(unsigned*)(p.y + obase + r) = pk.u;
    }
  }
}

// (1,32,32,128) f16 NHWC -> (128,32,32) f32 NCHW into d_out
__global__ __launch_bounds__(256) void finalize_k(const _Float16* h, float* out) {
  int t = blockIdx.x * 256 + threadIdx.x;
  if (t >= 128 * 32 * 32) return;
  int c = t >> 10;
  int pxy = t & 1023;
  out[t] = (float)h[(size_t)pxy * 128 + c];
}

// ---------------------------------------------------------------------------
// Host orchestration
// ---------------------------------------------------------------------------
// d_in leaf indices (setup_inputs dict insertion order, nested dicts in
// insertion order): vf, vc, batch_size, then params leaves.
enum {
  IN_VF = 0, IN_VC = 1, IN_BS = 2,
  W_INP = 3, BN_INP = 4,                               // inp_bn: g,be,mn,vr = 4..7
  W_C1B1_1 = 8,  B_C1B1_1 = 9,  BN_C1B1_1 = 10, W_C1B1_2 = 14, B_C1B1_2 = 15, BN_C1B1_2 = 16,
  W_C1B2_1 = 20, B_C1B2_1 = 21, BN_C1B2_1 = 22, W_C1B2_2 = 26, B_C1B2_2 = 27, BN_C1B2_2 = 28,
  W_C2DW = 32, BN_C2DW = 33,
  W_C2B1_1 = 37, B_C2B1_1 = 38, BN_C2B1_1 = 39, W_C2B1_2 = 43, B_C2B1_2 = 44, BN_C2B1_2 = 45,
  W_C2B2_1 = 49, B_C2B2_1 = 50, BN_C2B2_1 = 51, W_C2B2_2 = 55, B_C2B2_2 = 56, BN_C2B2_2 = 57,
  W_C3DW = 61, BN_C3DW = 62,
  W_C3B1_1 = 66, B_C3B1_1 = 67, BN_C3B1_1 = 68, W_C3B1_2 = 72, B_C3B1_2 = 73, BN_C3B1_2 = 74,
  W_C3B2_1 = 78, B_C3B2_1 = 79, BN_C3B2_1 = 80, W_C3B2_2 = 84, B_C3B2_2 = 85, BN_C3B2_2 = 86,
  W_C4DW = 90, BN_C4DW = 91,
  W_C4B1_1 = 95,  B_C4B1_1 = 96,  BN_C4B1_1 = 97,  W_C4B1_2 = 101, B_C4B1_2 = 102, BN_C4B1_2 = 103,
  W_C4B2_1 = 107, B_C4B2_1 = 108, BN_C4B2_1 = 109, W_C4B2_2 = 113, B_C4B2_2 = 114, BN_C4B2_2 = 115,
  W_MHA = 119, BN_MHA = 120
};

// conv-shape keys -> template instantiations
enum ConvKey { K_S1 = 0, K_C2DW, K_S2, K_C3DW, K_S3, K_C4DW, K_MHA };

template <int KD, int KH, int KW, int SX, int CI, int COT>
static void launchConvT(const ConvP& p, int OD, hipStream_t s) {
  constexpr int SPAN = (XT - 1) * SX + KW;
  long blocks = (long)OD * p.OH * (p.OW / 16);
  size_t smem = (size_t)KD * KH * SPAN * CI * 2;
  conv3d_wmma_k<KD, KH, KW, SX, CI, COT>
      <<<dim3((unsigned)blocks), dim3(32), smem, s>>>(p);
}

static void launchConv(int key, const ConvP& p, int OD, hipStream_t s) {
  switch (key) {
    case K_S1:   launchConvT<3, 3, 3, 1, 16, 1>(p, OD, s); break;
    case K_C2DW: launchConvT<3, 3, 3, 2, 16, 2>(p, OD, s); break;
    case K_S2:   launchConvT<3, 3, 3, 1, 32, 2>(p, OD, s); break;
    case K_C3DW: launchConvT<3, 3, 3, 2, 32, 4>(p, OD, s); break;
    case K_S3:   launchConvT<3, 3, 3, 1, 64, 4>(p, OD, s); break;
    case K_C4DW: launchConvT<3, 3, 3, 2, 64, 4>(p, OD, s); break;
    case K_MHA:  launchConvT<5, 3, 3, 1, 64, 8>(p, OD, s); break;
  }
}

extern "C" void kernel_launch(void* const* d_in, const int* in_sizes, int n_in,
                              void* d_out, int out_size, void* d_ws, size_t ws_size,
                              hipStream_t stream) {
  (void)n_in; (void)out_size; (void)ws_size;
  auto G = [&](int i) { return (const float*)d_in[i]; };
  const float* vf = (const float*)d_in[IN_VF];
  const int* vc = (const int*)d_in[IN_VC];

  char* ws = (char*)d_ws;
  size_t off = 0;
  auto alloc = [&](size_t bytes) -> void* {
    off = (off + 255) & ~(size_t)255;
    void* p = ws + off;
    off += bytes;
    return p;
  };

  const int D1 = 41, H1 = 256, W1 = 256;
  const int D2 = 21, H2 = 128, W2 = 128;
  const int D3 = 11, H3 = 64, W3 = 64;
  const int D4 = 5, H4 = 32, W4 = 32;
  const int D5 = 1, H5 = 32, W5 = 32;
  const long tot1 = (long)D1 * H1 * W1;

  float* accF = (float*)alloc((size_t)tot1 * 4 * sizeof(float));
  float* cntF = (float*)alloc((size_t)tot1 * sizeof(float));
  _Float16* bufA = (_Float16*)alloc((size_t)tot1 * 16 * 2);  // max activation: stage1, 16ch f16
  _Float16* bufB = (_Float16*)alloc((size_t)tot1 * 16 * 2);
  _Float16* m1 = (_Float16*)alloc((size_t)tot1 * 2);
  _Float16* m2 = (_Float16*)alloc((size_t)D2 * H2 * W2 * 2);
  _Float16* m3 = (_Float16*)alloc((size_t)D3 * H3 * W3 * 2);
  _Float16* m4 = (_Float16*)alloc((size_t)D4 * H4 * W4 * 2);
  _Float16* m5 = (_Float16*)alloc((size_t)D5 * H5 * W5 * 2);

  (void)hipMemsetAsync(accF, 0, (size_t)tot1 * 4 * sizeof(float), stream);
  (void)hipMemsetAsync(cntF, 0, (size_t)tot1 * sizeof(float), stream);

  const int NVn = in_sizes[IN_VF] / 4;
  scatter_k<<<dim3((NVn + 255) / 256), dim3(256), 0, stream>>>(vf, vc, accF, cntF, NVn, H1, W1);
  build_input_k<<<dim3((unsigned)((tot1 + 255) / 256)), dim3(256), 0, stream>>>(accF, cntF, bufA, m1, tot1);

  {
    long t2 = (long)D2 * H2 * W2, t3 = (long)D3 * H3 * W3, t4 = (long)D4 * H4 * W4, t5 = (long)D5 * H5 * W5;
    down_mask_k<<<dim3((unsigned)((t2 + 255) / 256)), dim3(256), 0, stream>>>(
        m1, m2, D1, H1, W1, D2, H2, W2, 3, 3, 3, 2, 2, 2, 1, 1, 1);
    down_mask_k<<<dim3((unsigned)((t3 + 255) / 256)), dim3(256), 0, stream>>>(
        m2, m3, D2, H2, W2, D3, H3, W3, 3, 3, 3, 2, 2, 2, 1, 1, 1);
    down_mask_k<<<dim3((unsigned)((t4 + 255) / 256)), dim3(256), 0, stream>>>(
        m3, m4, D3, H3, W3, D4, H4, W4, 3, 3, 3, 2, 2, 2, 0, 1, 1);
    down_mask_k<<<dim3((unsigned)((t5 + 255) / 256)), dim3(256), 0, stream>>>(
        m4, m5, D4, H4, W4, D5, H5, W5, 5, 3, 3, 5, 1, 1, 0, 1, 1);
  }

  auto convLayer = [&](int key, int wI, int bnI, int bI,
                       const _Float16* x, _Float16* y, const _Float16* resid, const _Float16* maskO,
                       int D, int H, int W, int OD, int OH, int OW,
                       int CiR, int Ci, int ciShift, int Co,
                       int KD, int KH, int KW, int sz, int sy, int pz, int py, int px) {
    int KT = KD * KH * KW;
    int Ksteps = (KT * Ci + 31) / 32;
    long tw = (long)(Co / 16) * Ksteps * 512;
    _Float16* wpk = (_Float16*)alloc((size_t)tw * 2);
    float* bias = (float*)alloc((size_t)Co * sizeof(float));

    PackP q;
    q.w = G(wI); q.g = G(bnI); q.vr = G(bnI + 3); q.wpk = wpk;
    q.CiR = CiR; q.Ci = Ci; q.ciShift = ciShift; q.Co = Co; q.KT = KT; q.Ksteps = Ksteps;
    pack_w_k<<<dim3((unsigned)((tw + 255) / 256)), dim3(256), 0, stream>>>(q, tw);
    pack_bias_k<<<dim3((Co + 63) / 64), dim3(64), 0, stream>>>(
        G(bnI), G(bnI + 1), G(bnI + 2), G(bnI + 3), bI >= 0 ? G(bI) : nullptr, bias, Co);

    ConvP p;
    p.x = x; p.wpk = wpk; p.bias = bias; p.maskO = maskO; p.resid = resid; p.y = y;
    p.D = D; p.H = H; p.W = W; p.OH = OH; p.OW = OW;
    p.sz = sz; p.sy = sy; p.pz = pz; p.py = py; p.px = px;
    launchConv(key, p, OD, stream);
  };

  // ---- stage 1 (16ch, full res) ----
  convLayer(K_S1, W_INP, BN_INP, -1, bufA, bufB, nullptr, m1,
            D1, H1, W1, D1, H1, W1, 4, 16, 4, 16, 3, 3, 3, 1, 1, 1, 1, 1);
  convLayer(K_S1, W_C1B1_1, BN_C1B1_1, B_C1B1_1, bufB, bufA, nullptr, m1,
            D1, H1, W1, D1, H1, W1, 16, 16, 4, 16, 3, 3, 3, 1, 1, 1, 1, 1);
  convLayer(K_S1, W_C1B1_2, BN_C1B1_2, B_C1B1_2, bufA, bufB, bufB, m1,
            D1, H1, W1, D1, H1, W1, 16, 16, 4, 16, 3, 3, 3, 1, 1, 1, 1, 1);
  convLayer(K_S1, W_C1B2_1, BN_C1B2_1, B_C1B2_1, bufB, bufA, nullptr, m1,
            D1, H1, W1, D1, H1, W1, 16, 16, 4, 16, 3, 3, 3, 1, 1, 1, 1, 1);
  convLayer(K_S1, W_C1B2_2, BN_C1B2_2, B_C1B2_2, bufA, bufB, bufB, m1,
            D1, H1, W1, D1, H1, W1, 16, 16, 4, 16, 3, 3, 3, 1, 1, 1, 1, 1);

  // ---- down to stage 2 (32ch) ----
  convLayer(K_C2DW, W_C2DW, BN_C2DW, -1, bufB, bufA, nullptr, m2,
            D1, H1, W1, D2, H2, W2, 16, 16, 4, 32, 3, 3, 3, 2, 2, 1, 1, 1);
  convLayer(K_S2, W_C2B1_1, BN_C2B1_1, B_C2B1_1, bufA, bufB, nullptr, m2,
            D2, H2, W2, D2, H2, W2, 32, 32, 5, 32, 3, 3, 3, 1, 1, 1, 1, 1);
  convLayer(K_S2, W_C2B1_2, BN_C2B1_2, B_C2B1_2, bufB, bufA, bufA, m2,
            D2, H2, W2, D2, H2, W2, 32, 32, 5, 32, 3, 3, 3, 1, 1, 1, 1, 1);
  convLayer(K_S2, W_C2B2_1, BN_C2B2_1, B_C2B2_1, bufA, bufB, nullptr, m2,
            D2, H2, W2, D2, H2, W2, 32, 32, 5, 32, 3, 3, 3, 1, 1, 1, 1, 1);
  convLayer(K_S2, W_C2B2_2, BN_C2B2_2, B_C2B2_2, bufB, bufA, bufA, m2,
            D2, H2, W2, D2, H2, W2, 32, 32, 5, 32, 3, 3, 3, 1, 1, 1, 1, 1);

  // ---- down to stage 3 (64ch) ----
  convLayer(K_C3DW, W_C3DW, BN_C3DW, -1, bufA, bufB, nullptr, m3,
            D2, H2, W2, D3, H3, W3, 32, 32, 5, 64, 3, 3, 3, 2, 2, 1, 1, 1);
  convLayer(K_S3, W_C3B1_1, BN_C3B1_1, B_C3B1_1, bufB, bufA, nullptr, m3,
            D3, H3, W3, D3, H3, W3, 64, 64, 6, 64, 3, 3, 3, 1, 1, 1, 1, 1);
  convLayer(K_S3, W_C3B1_2, BN_C3B1_2, B_C3B1_2, bufA, bufB, bufB, m3,
            D3, H3, W3, D3, H3, W3, 64, 64, 6, 64, 3, 3, 3, 1, 1, 1, 1, 1);
  convLayer(K_S3, W_C3B2_1, BN_C3B2_1, B_C3B2_1, bufB, bufA, nullptr, m3,
            D3, H3, W3, D3, H3, W3, 64, 64, 6, 64, 3, 3, 3, 1, 1, 1, 1, 1);
  convLayer(K_S3, W_C3B2_2, BN_C3B2_2, B_C3B2_2, bufA, bufB, bufB, m3,
            D3, H3, W3, D3, H3, W3, 64, 64, 6, 64, 3, 3, 3, 1, 1, 1, 1, 1);

  // ---- down to stage 4 (64ch, z pad 0) ----
  convLayer(K_C4DW, W_C4DW, BN_C4DW, -1, bufB, bufA, nullptr, m4,
            D3, H3, W3, D4, H4, W4, 64, 64, 6, 64, 3, 3, 3, 2, 2, 0, 1, 1);
  convLayer(K_S3, W_C4B1_1, BN_C4B1_1, B_C4B1_1, bufA, bufB, nullptr, m4,
            D4, H4, W4, D4, H4, W4, 64, 64, 6, 64, 3, 3, 3, 1, 1, 1, 1, 1);
  convLayer(K_S3, W_C4B1_2, BN_C4B1_2, B_C4B1_2, bufB, bufA, bufA, m4,
            D4, H4, W4, D4, H4, W4, 64, 64, 6, 64, 3, 3, 3, 1, 1, 1, 1, 1);
  convLayer(K_S3, W_C4B2_1, BN_C4B2_1, B_C4B2_1, bufA, bufB, nullptr, m4,
            D4, H4, W4, D4, H4, W4, 64, 64, 6, 64, 3, 3, 3, 1, 1, 1, 1, 1);
  convLayer(K_S3, W_C4B2_2, BN_C4B2_2, B_C4B2_2, bufB, bufA, bufA, m4,
            D4, H4, W4, D4, H4, W4, 64, 64, 6, 64, 3, 3, 3, 1, 1, 1, 1, 1);

  // ---- final (5,3,3)/(5,1,1) conv to 128ch, depth collapses to 1 ----
  convLayer(K_MHA, W_MHA, BN_MHA, -1, bufA, bufB, nullptr, m5,
            D4, H4, W4, D5, H5, W5, 64, 64, 6, 128, 5, 3, 3, 5, 1, 0, 1, 1);

  finalize_k<<<dim3((128 * 32 * 32 + 255) / 256), dim3(256), 0, stream>>>(bufB, (float*)d_out);
}
